// TokenPruner_35570919145562
// MI455X (gfx1250) — matrix operand order repaired and checked by reading, not compiled, not run on hardware
//
#include <hip/hip_runtime.h>
#include <hip/hip_bf16.h>
#include <stdint.h>

// Problem constants (fixed by reference setup_inputs: B=64,S=305,D=2048,c=1,keep=128)
#define B_    64
#define SEQ   305
#define D_    2048
#define P_    256     // NUM_PATCHES
#define T_    48      // task tokens = 305 - 1 - 256
#define TPAD  64      // task K padded to 2 WMMA K-steps
#define KEEP  128
#define OUTS  177     // 1 + 128 + 48
#define EPS_  1e-6f
#define SCALE 0.022097086912079612f  // 1/sqrt(2048)

typedef __attribute__((ext_vector_type(16))) __bf16 v16bf;
typedef __attribute__((ext_vector_type(8)))  float  v8f;

union FragU { uint4 q[2]; v16bf v; };

static __device__ inline unsigned short f2bf(float f) {
  union { float f; uint32_t u; } x; x.f = f;
  uint32_t r = x.u + 0x7FFFu + ((x.u >> 16) & 1u);   // round-to-nearest-even
  return (unsigned short)(r >> 16);
}

// Two 16B loads -> one 16xbf16 fragment (matches CDNA5 16-bit A/B VGPR striping)
static __device__ inline v16bf ldfrag(const unsigned short* p0, const unsigned short* p1) {
  FragU u;
  u.q[0] = *(const uint4*)p0;
  u.q[1] = *(const uint4*)p1;
  return u.v;
}

static __device__ inline v8f wmma_bf16(v16bf a, v16bf b, v8f c) {
  return __builtin_amdgcn_wmma_f32_16x16x32_bf16(false, a, false, b, (short)0, c, false, false);
}

// ---------------------------------------------------------------- kernel 0: zero
__global__ void k_zero(uint32_t* __restrict__ tnt, uint32_t* __restrict__ counts) {
  size_t n1 = (size_t)B_ * D_ * TPAD / 2;  // tnt as dwords
  size_t n2 = (size_t)B_ * P_;
  size_t i = (size_t)blockIdx.x * blockDim.x + threadIdx.x;
  size_t stride = (size_t)gridDim.x * blockDim.x;
  for (size_t k = i; k < n1; k += stride) tnt[k] = 0u;
  for (size_t k = i; k < n2; k += stride) counts[k] = 0u;
}

// -------------------------------------------------- kernel 1: RMS-norm -> bf16
// One 256-thread block per normalized row (patches then task tokens).
// Task rows also scattered into transposed layout tnt[b][d][t] (K-contiguous B
// operand for the queries GEMM).
__global__ void k_rmsnorm(const float* __restrict__ tokens,
                          unsigned short* __restrict__ pn,
                          unsigned short* __restrict__ tn,
                          unsigned short* __restrict__ tnt) {
  int blk = blockIdx.x;
  int b = blk / (P_ + T_);
  int w = blk % (P_ + T_);
  int srow = (w < P_) ? (1 + w) : (P_ + 1 + (w - P_));
  const float* src = tokens + ((size_t)b * SEQ + srow) * D_;
  int tid = threadIdx.x;
  float4 a = ((const float4*)src)[tid];
  float4 c = ((const float4*)src)[tid + 256];
  float ss = a.x*a.x + a.y*a.y + a.z*a.z + a.w*a.w
           + c.x*c.x + c.y*c.y + c.z*c.z + c.w*c.w;
  __shared__ float red[256];
  red[tid] = ss;
  __syncthreads();
  for (int s = 128; s > 0; s >>= 1) {
    if (tid < s) red[tid] += red[tid + s];
    __syncthreads();
  }
  float rr = rsqrtf(red[0] * (1.0f / D_) + EPS_);
  float v0 = a.x*rr, v1 = a.y*rr, v2 = a.z*rr, v3 = a.w*rr;
  float v4 = c.x*rr, v5 = c.y*rr, v6 = c.z*rr, v7 = c.w*rr;
  uint2 p, q;
  p.x = (uint32_t)f2bf(v0) | ((uint32_t)f2bf(v1) << 16);
  p.y = (uint32_t)f2bf(v2) | ((uint32_t)f2bf(v3) << 16);
  q.x = (uint32_t)f2bf(v4) | ((uint32_t)f2bf(v5) << 16);
  q.y = (uint32_t)f2bf(v6) | ((uint32_t)f2bf(v7) << 16);
  if (w < P_) {
    unsigned short* dst = pn + ((size_t)b * P_ + w) * D_;
    ((uint2*)dst)[tid]       = p;
    ((uint2*)dst)[tid + 256] = q;
  } else {
    int t = w - P_;
    unsigned short* dst = tn + ((size_t)b * T_ + t) * D_;
    ((uint2*)dst)[tid]       = p;
    ((uint2*)dst)[tid + 256] = q;
    unsigned short bv[8] = { f2bf(v0), f2bf(v1), f2bf(v2), f2bf(v3),
                             f2bf(v4), f2bf(v5), f2bf(v6), f2bf(v7) };
    int d0 = tid * 4;
    int d1 = 1024 + tid * 4;
    unsigned short* tb = tnt + (size_t)b * D_ * TPAD + t;
    for (int k = 0; k < 4; ++k) tb[(size_t)(d0 + k) * TPAD] = bv[k];
    for (int k = 0; k < 4; ++k) tb[(size_t)(d1 + k) * TPAD] = bv[4 + k];
  }
}

// --------------------------- kernel 2: logits = pn.tn^T * scale, softmax -> attn
// One wave per (batch, 16-patch M-tile). 3 N-tiles cover the 48 task tokens.
// D-tile mapping: M = v + 8*half, N = lane&15; softmax reduces across the
// 16-lane half-group holding each row via shfl_xor (masks 1..8 stay in-half).
__global__ void k_attn(const unsigned short* __restrict__ pn,
                       const unsigned short* __restrict__ tn,
                       unsigned short* __restrict__ attn) {
  int b  = blockIdx.x >> 4;
  int m0 = (blockIdx.x & 15) << 4;
  int lane = threadIdx.x;
  int h  = lane >> 4;
  int ln = lane & 15;
  const unsigned short* arow = pn + ((size_t)b * P_ + m0 + ln) * D_;
  const unsigned short* b0r = tn + ((size_t)b * T_ + ln) * D_;
  const unsigned short* b1r = b0r + (size_t)16 * D_;
  const unsigned short* b2r = b0r + (size_t)32 * D_;
  v8f acc0 = {}, acc1 = {}, acc2 = {};
  for (int kb = 0; kb < D_; kb += 32) {
    v16bf a = ldfrag(arow + kb + 8*h, arow + kb + 16 + 8*h);
    acc0 = wmma_bf16(a, ldfrag(b0r + kb + 8*h, b0r + kb + 16 + 8*h), acc0);
    acc1 = wmma_bf16(a, ldfrag(b1r + kb + 8*h, b1r + kb + 16 + 8*h), acc1);
    acc2 = wmma_bf16(a, ldfrag(b2r + kb + 8*h, b2r + kb + 16 + 8*h), acc2);
  }
  unsigned short* orow = attn + ((size_t)b * P_ + m0) * TPAD;
  for (int v = 0; v < 8; ++v) {
    float t0 = acc0[v] * SCALE, t1 = acc1[v] * SCALE, t2 = acc2[v] * SCALE;
    float mx = fmaxf(t0, fmaxf(t1, t2));
    for (int m = 1; m <= 8; m <<= 1) mx = fmaxf(mx, __shfl_xor(mx, m, 32));
    float e0 = expf(t0 - mx), e1 = expf(t1 - mx), e2 = expf(t2 - mx);
    float s = e0 + e1 + e2;
    for (int m = 1; m <= 8; m <<= 1) s += __shfl_xor(s, m, 32);
    float r = 1.0f / s;
    unsigned short* rp = orow + (size_t)(v + 8*h) * TPAD;
    rp[ln]      = f2bf(e0 * r);
    rp[16 + ln] = f2bf(e1 * r);
    rp[32 + ln] = f2bf(e2 * r);
    rp[48 + ln] = 0;                 // K-pad so the queries GEMM runs clean K=64
  }
}

// ------------------- kernel 3: queries = rmsnorm(attn . tn) -> q (bf16)
// 8 waves per (batch, 16-patch tile); each wave owns 256 of the 2048 d's.
// Two passes over N-tiles: pass1 accumulates row sum-of-squares (LDS reduce),
// pass2 recomputes (K=64 -> only 2 WMMAs/tile, cheap) and writes scaled bf16.
__global__ void k_queries(const unsigned short* __restrict__ attn,
                          const unsigned short* __restrict__ tnt,
                          unsigned short* __restrict__ q) {
  int b  = blockIdx.x >> 4;
  int m0 = (blockIdx.x & 15) << 4;
  int tid = threadIdx.x;
  int wave = tid >> 5, lane = tid & 31;
  int h = lane >> 4, ln = lane & 15;
  const unsigned short* arow = attn + ((size_t)b * P_ + m0 + ln) * TPAD;
  v16bf a0 = ldfrag(arow + 8*h,      arow + 16 + 8*h);       // K 0..31
  v16bf a1 = ldfrag(arow + 32 + 8*h, arow + 48 + 8*h);       // K 32..63
  const unsigned short* tb = tnt + (size_t)b * D_ * TPAD;
  __shared__ float ssq[16][8];
  __shared__ float rrms[16];
  float ssqv[8];
  for (int v = 0; v < 8; ++v) ssqv[v] = 0.f;
  for (int i = 0; i < 16; ++i) {                 // pass 1
    int d0 = (wave * 16 + i) * 16;
    const unsigned short* cp = tb + (size_t)(d0 + ln) * TPAD + 8*h;
    v8f acc = {};
    acc = wmma_bf16(a0, ldfrag(cp,      cp + 16), acc);
    acc = wmma_bf16(a1, ldfrag(cp + 32, cp + 48), acc);
    for (int v = 0; v < 8; ++v) ssqv[v] += acc[v] * acc[v];
  }
  for (int v = 0; v < 8; ++v) {
    float s = ssqv[v];
    for (int m = 1; m <= 8; m <<= 1) s += __shfl_xor(s, m, 32);
    if (ln == 0) ssq[v + 8*h][wave] = s;
  }
  __syncthreads();
  if (tid < 16) {
    float t = 0.f;
    for (int wv = 0; wv < 8; ++wv) t += ssq[tid][wv];
    rrms[tid] = rsqrtf(t * (1.0f / D_) + EPS_);
  }
  __syncthreads();
  float rv[8];
  for (int v = 0; v < 8; ++v) rv[v] = rrms[v + 8*h];
  unsigned short* qb = q + ((size_t)b * P_ + m0) * D_;
  for (int i = 0; i < 16; ++i) {                 // pass 2
    int d0 = (wave * 16 + i) * 16;
    const unsigned short* cp = tb + (size_t)(d0 + ln) * TPAD + 8*h;
    v8f acc = {};
    acc = wmma_bf16(a0, ldfrag(cp,      cp + 16), acc);
    acc = wmma_bf16(a1, ldfrag(cp + 32, cp + 48), acc);
    for (int v = 0; v < 8; ++v)
      qb[(size_t)(v + 8*h) * D_ + d0 + ln] = f2bf(acc[v] * rv[v]);
  }
}

// ---------- kernel 4: score = q . pn^T, row argmax (lowest-index tie), counts
// Positive scale dropped: argmax is invariant. 8 waves x 32 cols cover N=256.
__global__ void k_score(const unsigned short* __restrict__ q,
                        const unsigned short* __restrict__ pn,
                        int* __restrict__ counts) {
  int b  = blockIdx.x >> 4;
  int m0 = (blockIdx.x & 15) << 4;
  int tid = threadIdx.x;
  int wave = tid >> 5, lane = tid & 31;
  int h = lane >> 4, ln = lane & 15;
  const unsigned short* arow = q + ((size_t)b * P_ + m0 + ln) * D_;
  const unsigned short* b0r = pn + ((size_t)b * P_ + wave * 32 + ln) * D_;
  const unsigned short* b1r = b0r + (size_t)16 * D_;
  v8f acc0 = {}, acc1 = {};
  for (int kb = 0; kb < D_; kb += 32) {
    v16bf a = ldfrag(arow + kb + 8*h, arow + kb + 16 + 8*h);
    acc0 = wmma_bf16(a, ldfrag(b0r + kb + 8*h, b0r + kb + 16 + 8*h), acc0);
    acc1 = wmma_bf16(a, ldfrag(b1r + kb + 8*h, b1r + kb + 16 + 8*h), acc1);
  }
  __shared__ float bvv[16][8];
  __shared__ int   biv[16][8];
  int base = wave * 32;
  for (int v = 0; v < 8; ++v) {
    float bv = acc0[v]; int bi = base + ln;
    float o  = acc1[v]; int oi = base + 16 + ln;
    if (o > bv) { bv = o; bi = oi; }             // oi > bi: ties keep lower idx
    for (int m = 1; m <= 8; m <<= 1) {
      float ov = __shfl_xor(bv, m, 32);
      int   ox = __shfl_xor(bi, m, 32);
      if (ov > bv || (ov == bv && ox < bi)) { bv = ov; bi = ox; }
    }
    if (ln == 0) { bvv[v + 8*h][wave] = bv; biv[v + 8*h][wave] = bi; }
  }
  __syncthreads();
  if (tid < 16) {
    float bv = bvv[tid][0]; int bi = biv[tid][0];
    for (int wv = 1; wv < 8; ++wv) {             // cols ascend with wave: ">" keeps lowest idx
      if (bvv[tid][wv] > bv) { bv = bvv[tid][wv]; bi = biv[tid][wv]; }
    }
    atomicAdd(&counts[b * P_ + bi], 1);
  }
}

// --------------- kernel 5: top-128 by count (ties -> lower index), orig order
// rank_i = #{j: c_j > c_i} + #{j<i: c_j == c_i}; keep iff rank < 128. This is
// exactly jax top_k + stable argsort of the keep mask.
__global__ void k_select(const int* __restrict__ counts, int* __restrict__ order) {
  int b = blockIdx.x;
  int i = threadIdx.x;
  __shared__ int cnt[P_];
  __shared__ unsigned char keep[P_];
  cnt[i] = counts[b * P_ + i];
  __syncthreads();
  int ci = cnt[i];
  int rank = 0;
  for (int j = 0; j < P_; ++j) {
    int cj = cnt[j];
    rank += (cj > ci) || (cj == ci && j < i);
  }
  keep[i] = (rank < KEEP) ? 1 : 0;
  __syncthreads();
  int before = 0;
  for (int j = 0; j < i; ++j) before += keep[j];
  if (keep[i]) order[b * KEEP + before] = i;
}

// ---------------------------- kernel 6: gather [cls | kept patches | task]
__global__ void k_gather(const float* __restrict__ tokens,
                         const int* __restrict__ pos_ids,
                         const int* __restrict__ amask,
                         const int* __restrict__ order,
                         float* __restrict__ out_tokens,
                         float* __restrict__ out_pos,
                         float* __restrict__ out_mask) {
  int b = blockIdx.x / OUTS;
  int r = blockIdx.x % OUTS;
  int s;
  if (r == 0)          s = 0;                          // cls
  else if (r < 1+KEEP) s = 1 + order[b * KEEP + (r - 1)];
  else                 s = 1 + P_ + (r - (1 + KEEP));  // task
  const float* src = tokens + ((size_t)b * SEQ + s) * D_;
  float* dst = out_tokens + ((size_t)b * OUTS + r) * D_;
  int t = threadIdx.x;
  ((float4*)dst)[t]       = ((const float4*)src)[t];
  ((float4*)dst)[t + 256] = ((const float4*)src)[t + 256];
  if (t == 0) {
    out_pos [(size_t)b * OUTS + r] = (float)pos_ids[b * SEQ + s];
    out_mask[(size_t)b * OUTS + r] = (float)amask [b * SEQ + s];
  }
}

extern "C" void kernel_launch(void* const* d_in, const int* in_sizes, int n_in,
                              void* d_out, int out_size, void* d_ws, size_t ws_size,
                              hipStream_t stream) {
  (void)in_sizes; (void)n_in; (void)out_size; (void)ws_size;
  const float* tokens = (const float*)d_in[0];
  const int*   pos_ids = (const int*)d_in[1];
  const int*   amask   = (const int*)d_in[2];
  // d_in[3] = cls_token_count (1), d_in[4] = keep_tokens (128): fixed by setup.

  char* ws = (char*)d_ws;
  size_t off = 0;
  unsigned short* pn   = (unsigned short*)(ws + off); off += (size_t)B_ * P_ * D_ * 2;
  unsigned short* tn   = (unsigned short*)(ws + off); off += (size_t)B_ * T_ * D_ * 2;
  unsigned short* tnt  = (unsigned short*)(ws + off); off += (size_t)B_ * D_ * TPAD * 2;
  unsigned short* attn = (unsigned short*)(ws + off); off += (size_t)B_ * P_ * TPAD * 2;
  unsigned short* q    = (unsigned short*)(ws + off); off += (size_t)B_ * P_ * D_ * 2;
  int* counts = (int*)(ws + off); off += (size_t)B_ * P_ * 4;
  int* order  = (int*)(ws + off); off += (size_t)B_ * KEEP * 4;

  float* out_tokens = (float*)d_out;
  float* out_pos  = out_tokens + (size_t)B_ * OUTS * D_;
  float* out_mask = out_pos    + (size_t)B_ * OUTS;

  k_zero   <<<4096, 256, 0, stream>>>((uint32_t*)tnt, (uint32_t*)counts);
  k_rmsnorm<<<B_ * (P_ + T_), 256, 0, stream>>>(tokens, pn, tn, tnt);
  k_attn   <<<B_ * 16, 32, 0, stream>>>(pn, tn, attn);
  k_queries<<<B_ * 16, 256, 0, stream>>>(attn, tnt, q);
  k_score  <<<B_ * 16, 256, 0, stream>>>(q, pn, counts);
  k_select <<<B_, 256, 0, stream>>>(counts, order);
  k_gather <<<B_ * OUTS, 256, 0, stream>>>(tokens, pos_ids, amask, order,
                                           out_tokens, out_pos, out_mask);
}
